// WordLevelMHA_65403761984241
// MI455X (gfx1250) — matrix operand order repaired and checked by reading, not compiled, not run on hardware
//
#include <hip/hip_runtime.h>

#define E  1024
#define NC 8
#define CD 128
#define H  2
#define QK 32
#define VD 64
#define NW 7               // waves per block
#define TPB (NW * 32)
#define PW_BYTES 29696     // per-wave LDS scratch
#define FIXED_BYTES 102400 // PE + bf16 weights

typedef __bf16 bf16_t;
typedef bf16_t v16bf __attribute__((ext_vector_type(16)));
typedef float  v8f   __attribute__((ext_vector_type(8)));

__device__ __forceinline__ unsigned short f2bf(float f) {
    unsigned u = __float_as_uint(f);
    u += 0x7FFFu + ((u >> 16) & 1u);       // round-to-nearest-even
    return (unsigned short)(u >> 16);
}
__device__ __forceinline__ float bf2f(unsigned short s) {
    return __uint_as_float(((unsigned)s) << 16);
}

// generic shared pointer -> 32-bit LDS byte offset (addrspace(3) ptrtoint)
__device__ __forceinline__ unsigned lds_off(const void* p) {
    return (unsigned)(unsigned long long)(__attribute__((address_space(3))) const char*)(const char*)p;
}

union FragU { uint4 u[2]; v16bf v; };

// A-matrix fragment 16x32 bf16 from LDS [16][rs] row-major
__device__ __forceinline__ v16bf load_fragA(const unsigned short* base, int rs,
                                            int l16, int half, int k0) {
    FragU U;
    const unsigned short* p = base + l16 * rs + k0 + half * 16;
    U.u[0] = *(const uint4*)(p);       // K = k0+half*16   .. +7
    U.u[1] = *(const uint4*)(p + 8);   // K = k0+half*16+8 .. +15
    return U.v;
}
// B-matrix fragment 32x16 bf16 from LDS stored as [N][K] row-major
__device__ __forceinline__ v16bf load_fragB(const unsigned short* base, int rs,
                                            int l16, int half, int n0, int k0) {
    FragU U;
    const uint4* p = (const uint4*)(base + (n0 + l16) * rs + k0 + half * 16);
    U.u[0] = p[0];
    U.u[1] = p[1];
    return U.v;
}

__global__ void __launch_bounds__(TPB, 1)
mha_wmma_kernel(const float* __restrict__ x,
                const float* __restrict__ Wq, const float* __restrict__ Wk,
                const float* __restrict__ Wv, const float* __restrict__ Wo,
                const float* __restrict__ gamma, const float* __restrict__ beta,
                const float* __restrict__ rel_bias,
                float* __restrict__ y, float* __restrict__ attn_g,
                long long tiles)
{
    extern __shared__ char smem[];
    float*          pe = (float*)smem;                       // 8*128 f32     (4 KB)
    unsigned short* wq = (unsigned short*)(smem + 4096);     // 64*128 bf16
    unsigned short* wk = wq + 64 * 128;
    unsigned short* wv = wk + 64 * 128;                      // 128*128 bf16
    unsigned short* wo = wv + 128 * 128;
    char* wavebase = smem + FIXED_BYTES;

    const int tid = threadIdx.x;

    // ---- block-cooperative staging: PE table + bf16 weights ----
    for (int i = tid; i < NC * CD; i += TPB) {
        int n = i >> 7, c = i & 127;
        float div = __expf((float)(c & ~1) * (-9.210340371976184f / (float)CD));
        float ang = (float)n * div;
        pe[i] = (c & 1) ? __cosf(ang) : __sinf(ang);
    }
    for (int i = tid; i < 64 * 128; i += TPB)  { wq[i] = f2bf(Wq[i]); wk[i] = f2bf(Wk[i]); }
    for (int i = tid; i < 128 * 128; i += TPB) { wv[i] = f2bf(Wv[i]); wo[i] = f2bf(Wo[i]); }
    __syncthreads();

    const int w    = tid >> 5;
    const int lane = tid & 31;
    const int l16  = lane & 15;
    const int half = lane >> 4;

    // per-wave LDS scratch (29696 B)
    float*          stg  = (float*)(wavebase + (size_t)w * PW_BYTES);  // 2 x (16x128 f32): raw x, then h in-place
    unsigned short* xcb  = (unsigned short*)((char*)stg + 16384);      // 16x128 bf16 (xc, then ctx)
    unsigned short* qb   = xcb + 16 * 128;                             // 16x64 bf16
    unsigned short* kb   = qb + 16 * 64;                               // 16x64 bf16
    unsigned short* vb   = kb + 16 * 64;                               // 16x128 bf16
    float*          att  = (float*)((char*)vb + 16 * 128 * 2);         // 16x16 f32

    const unsigned stg_off = lds_off(stg);
    const long long wgid   = (long long)blockIdx.x * NW + w;
    const long long stride = (long long)gridDim.x * NW;

    // ---- prologue: async-prefetch first tile's raw x into stage 0 ----
    if (wgid < tiles) {
        #pragma unroll
        for (int r = 0; r < 16; ++r) {
            long long b = wgid * 2 + (r >> 3);
            unsigned long long ga = (unsigned long long)(x + b * E + (r & 7) * CD + lane * 4);
            unsigned la = stg_off + r * 512 + lane * 16;
            asm volatile("global_load_async_to_lds_b128 %0, %1, off"
                         :: "v"(la), "v"(ga) : "memory");
        }
    }

    const float scale = 0.17677669529663687f;   // 1/sqrt(32)
    int buf = 0;

    for (long long t = wgid; t < tiles; t += stride) {
        const long long b0 = t * 2;
        float* st = stg + buf * (16 * 128);

        // wait for this tile's async copy, then prefetch next tile into other buffer
        asm volatile("s_wait_asynccnt 0x0" ::: "memory");
        const long long tn = t + stride;
        if (tn < tiles) {
            unsigned nla = stg_off + (buf ^ 1) * 8192;
            #pragma unroll
            for (int r = 0; r < 16; ++r) {
                long long b = tn * 2 + (r >> 3);
                unsigned long long ga = (unsigned long long)(x + b * E + (r & 7) * CD + lane * 4);
                unsigned la = nla + r * 512 + lane * 16;
                asm volatile("global_load_async_to_lds_b128 %0, %1, off"
                             :: "v"(la), "v"(ga) : "memory");
            }
        }

        // ---- build bf16 xc tile (x + PE) ----
        for (int r = 0; r < 16; ++r) {
            const float4 xv = *(const float4*)(st + r * 128 + lane * 4);
            const float4 pv = *(const float4*)(pe + (r & 7) * 128 + lane * 4);
            uint2 hb;
            hb.x = (unsigned)f2bf(xv.x + pv.x) | ((unsigned)f2bf(xv.y + pv.y) << 16);
            hb.y = (unsigned)f2bf(xv.z + pv.z) | ((unsigned)f2bf(xv.w + pv.w) << 16);
            *(uint2*)(xcb + r * 128 + lane * 4) = hb;
        }

        // ---- Q and K projections: [16x128] x [128x64] via WMMA bf16 ----
        for (int tt = 0; tt < 4; ++tt) {
            v8f aq = {0, 0, 0, 0, 0, 0, 0, 0};
            v8f ak = {0, 0, 0, 0, 0, 0, 0, 0};
            for (int kk = 0; kk < 128; kk += 32) {
                v16bf a  = load_fragA(xcb, 128, l16, half, kk);
                v16bf bq = load_fragB(wq, 128, l16, half, tt * 16, kk);
                aq = __builtin_amdgcn_wmma_f32_16x16x32_bf16(false, a, false, bq, (short)0, aq, false, false);
                v16bf bk = load_fragB(wk, 128, l16, half, tt * 16, kk);
                ak = __builtin_amdgcn_wmma_f32_16x16x32_bf16(false, a, false, bk, (short)0, ak, false, false);
            }
            #pragma unroll
            for (int v = 0; v < 8; ++v) {
                qb[(v + half * 8) * 64 + tt * 16 + l16] = f2bf(aq[v]);
                kb[(v + half * 8) * 64 + tt * 16 + l16] = f2bf(ak[v]);
            }
        }
        // ---- V projection: [16x128] x [128x128] ----
        for (int tt = 0; tt < 8; ++tt) {
            v8f av = {0, 0, 0, 0, 0, 0, 0, 0};
            for (int kk = 0; kk < 128; kk += 32) {
                v16bf a  = load_fragA(xcb, 128, l16, half, kk);
                v16bf bv = load_fragB(wv, 128, l16, half, tt * 16, kk);
                av = __builtin_amdgcn_wmma_f32_16x16x32_bf16(false, a, false, bv, (short)0, av, false, false);
            }
            #pragma unroll
            for (int v = 0; v < 8; ++v)
                vb[(v + half * 8) * 128 + tt * 16 + l16] = f2bf(av[v]);
        }

        // ---- attention per head (scores via WMMA; xcb reused as ctx bf16) ----
        for (int h = 0; h < H; ++h) {
            v8f s = {0, 0, 0, 0, 0, 0, 0, 0};
            v16bf aQ = load_fragA(qb, 64, l16, half, h * QK);
            v16bf bK = load_fragB(kb, 64, l16, half, 0, h * QK);
            s = __builtin_amdgcn_wmma_f32_16x16x32_bf16(false, aQ, false, bK, (short)0, s, false, false);

            // valid lanes: 0..7 (batch b0, m=lane) and 24..31 (batch b0+1, m=l16-8)
            int m = (lane < 16) ? l16 : (l16 - 8);
            float p[8];
            #pragma unroll
            for (int v = 0; v < 8; ++v) {
                int bi = m - v + 7;
                bi = bi < 0 ? 0 : (bi > 14 ? 14 : bi);
                p[v] = s[v] * scale + rel_bias[h * 15 + bi];
            }
            #pragma unroll
            for (int v = 0; v < 8; ++v) {       // softmax over aligned 8-lane groups
                float mx = p[v];
                mx = fmaxf(mx, __shfl_xor(mx, 1, 32));
                mx = fmaxf(mx, __shfl_xor(mx, 2, 32));
                mx = fmaxf(mx, __shfl_xor(mx, 4, 32));
                float e = __expf(p[v] - mx);
                float sm = e;
                sm += __shfl_xor(sm, 1, 32);
                sm += __shfl_xor(sm, 2, 32);
                sm += __shfl_xor(sm, 4, 32);
                p[v] = e / sm;
            }
            if (lane < 8) {
                #pragma unroll
                for (int v = 0; v < 8; ++v) {
                    att[v * 16 + lane] = p[v];
                    attn_g[((b0 * H + h) * NC + v) * NC + lane] = p[v];
                }
            } else if (lane >= 24) {
                #pragma unroll
                for (int v = 0; v < 8; ++v) {
                    att[(v + 8) * 16 + l16] = p[v];
                    attn_g[(((b0 + 1) * H + h) * NC + v) * NC + (l16 - 8)] = p[v];
                }
            }

            // ctx[n, h*64 + j] = sum_m attn[n,m] * V[m, h*64+j]  (K=8: VALU)
            float cacc[32];
            #pragma unroll
            for (int j = 0; j < 32; ++j) cacc[j] = 0.f;
            const int vrow0 = (l16 < 8) ? 0 : 8;
            for (int mm = 0; mm < 8; ++mm) {
                float aw = att[l16 * 16 + vrow0 + mm];
                const unsigned short* vr = vb + (vrow0 + mm) * 128 + h * 64 + half * 32;
                #pragma unroll
                for (int j = 0; j < 32; ++j)
                    cacc[j] = fmaf(aw, bf2f(vr[j]), cacc[j]);
            }
            unsigned short* cr = xcb + l16 * 128 + h * 64 + half * 32;   // ctx in bf16
            #pragma unroll
            for (int j = 0; j < 32; ++j) cr[j] = f2bf(cacc[j]);
        }

        // ---- output projection + residual: h = (x + PE) + out, in place over stage ----
        for (int tt = 0; tt < 8; ++tt) {
            v8f acc = {0, 0, 0, 0, 0, 0, 0, 0};
            for (int kk = 0; kk < 128; kk += 32) {
                v16bf a  = load_fragA(xcb, 128, l16, half, kk);
                v16bf bw = load_fragB(wo, 128, l16, half, tt * 16, kk);
                acc = __builtin_amdgcn_wmma_f32_16x16x32_bf16(false, a, false, bw, (short)0, acc, false, false);
            }
            #pragma unroll
            for (int v = 0; v < 8; ++v) {
                int row = v + half * 8;
                int col = tt * 16 + l16;
                int idx = row * 128 + col;
                st[idx] = st[idx] + pe[(row & 7) * 128 + col] + acc[v];
            }
        }

        // ---- LayerNorm: row r=l16, cols half*64..+63; partner reduce via shfl_xor 16 ----
        {
            const float* hr = st + l16 * 128;
            float s1 = 0.f, s2 = 0.f;
            for (int j = 0; j < 64; ++j) {
                float v = hr[half * 64 + j];
                s1 += v; s2 += v * v;
            }
            s1 += __shfl_xor(s1, 16, 32);
            s2 += __shfl_xor(s2, 16, 32);
            float mu  = s1 * (1.f / 128.f);
            float var = s2 * (1.f / 128.f) - mu * mu;
            float rs  = rsqrtf(var + 1e-5f);
            long long b = b0 + (l16 >> 3);
            int n = l16 & 7;
            float* yr = y + b * E + n * CD + half * 64;
            for (int j = 0; j < 64; ++j) {
                int c = half * 64 + j;
                yr[j] = (hr[c] - mu) * rs * gamma[c] + beta[c];
            }
        }

        buf ^= 1;
    }
}

extern "C" void kernel_launch(void* const* d_in, const int* in_sizes, int n_in,
                              void* d_out, int out_size, void* d_ws, size_t ws_size,
                              hipStream_t stream) {
    const float* x        = (const float*)d_in[0];
    const float* Wq       = (const float*)d_in[1];
    const float* Wk       = (const float*)d_in[2];
    const float* Wv       = (const float*)d_in[3];
    const float* Wo       = (const float*)d_in[4];
    const float* gamma    = (const float*)d_in[5];
    const float* beta     = (const float*)d_in[6];
    const float* rel_bias = (const float*)d_in[7];

    float* y = (float*)d_out;
    long long Bn = (long long)in_sizes[0] / E;      // 65536
    float* attn = y + Bn * E;                       // outputs concatenated: y then attn

    long long tiles = Bn / 2;                       // 16 rows (2 batch) per tile
    int blocks = 1024;                              // persistent; grid-stride over tiles
    long long maxb = (tiles + NW - 1) / NW;
    if ((long long)blocks > maxb) blocks = (int)maxb;

    size_t shmem = FIXED_BYTES + (size_t)NW * PW_BYTES;   // 310272 B

    hipLaunchKernelGGL(mha_wmma_kernel, dim3(blocks), dim3(TPB), shmem, stream,
                       x, Wq, Wk, Wv, Wo, gamma, beta, rel_bias, y, attn, tiles);
}
